// DeepGATv2_88862873355007
// MI455X (gfx1250) — matrix operand order repaired and checked by reading, not compiled, not run on hardware
//
#include <hip/hip_runtime.h>
#include <hip/hip_bf16.h>
#include <math.h>

// ---------------- problem constants (from reference) ----------------
#define NN    100000
#define EE    1200000
#define ET    (EE + NN)      // edges incl. self loops
#define BB    256
#define INC   12
#define HC    64
#define OUTC  58
#define SLOPE 0.2f

typedef _Float16 v16h __attribute__((ext_vector_type(16)));
typedef _Float16 v8h  __attribute__((ext_vector_type(8)));
typedef float    v8f  __attribute__((ext_vector_type(8)));

// Order-preserving float -> u32 map for atomic segment-max (deterministic).
__device__ __forceinline__ unsigned encMax(float f) {
    int i = __float_as_int(f);
    return (i >= 0) ? ((unsigned)i | 0x80000000u) : ~(unsigned)i;
}
__device__ __forceinline__ float decMax(unsigned u) {
    int i = (u & 0x80000000u) ? (int)(u & 0x7fffffffu) : (int)(~u);
    return __int_as_float(i);
}

__global__ void k_zero(float* __restrict__ p, int n) {
    int i = blockIdx.x * blockDim.x + threadIdx.x;
    if (i < n) p[i] = 0.f;
}

__global__ void k_pad_att(const float* __restrict__ att, float* __restrict__ attp, int C) {
    int c = threadIdx.x;
    if (c < 64) attp[c] = (c < C) ? att[c] : 0.f;
}

// Convert layer-0 input (N x 12 f32) into zero-padded N x 64 f16 staging buffer.
__global__ void k_conv_x0(const float* __restrict__ x, _Float16* __restrict__ Xh) {
    int i = blockIdx.x * blockDim.x + threadIdx.x;
    if (i >= NN * 64) return;
    int n = i >> 6, k = i & 63;
    float v = (k < INC) ? x[(size_t)n * INC + k] : 0.f;
    Xh[i] = (_Float16)v;
}

// Convert + transpose W (Cin x Cout f32, row-major) into 64 x 64 f16 (col-major
// in K), zero-padded, so each B fragment is a contiguous K-run per column.
__global__ void k_conv_w(const float* __restrict__ Wl, const float* __restrict__ Wr,
                         _Float16* __restrict__ WlT, _Float16* __restrict__ WrT,
                         int Cin, int Cout) {
    int i = blockIdx.x * blockDim.x + threadIdx.x;
    if (i >= 2 * 64 * 64) return;
    int m = i >> 12, c = (i >> 6) & 63, k = i & 63;
    const float* W = m ? Wr : Wl;
    _Float16*    T = m ? WrT : WlT;
    float v = (k < Cin && c < Cout) ? W[(size_t)k * Cout + c] : 0.f;
    T[(size_t)c * 64 + k] = (_Float16)v;
}

// ---------------- WMMA node transform: Y{l,r} = X @ W{l,r} ----------------
// One wave per 16-row stripe; computes all 4 col-tiles for BOTH Wl and Wr
// (8 f32 accumulators), reusing the A fragment 8x. All operand loads are
// unconditional b128 loads from the padded f16 staging buffers.
__global__ void gat_transform(const _Float16* __restrict__ Xh,
                              const _Float16* __restrict__ WlT,
                              const _Float16* __restrict__ WrT,
                              float* __restrict__ Yl, float* __restrict__ Yr,
                              int ksteps, int rowTiles)
{
    int wid  = blockIdx.x * (blockDim.x >> 5) + (threadIdx.x >> 5);
    int lane = threadIdx.x & 31;
    if (wid >= rowTiles) return;        // wave-uniform: EXEC all-1s at every WMMA

    int l15 = lane & 15;
    int hi  = lane >> 4;
    int row = wid * 16 + l15;
    const v8h* pA = (const v8h*)(Xh + (size_t)row * 64);

    v8f acc[8];
#pragma unroll
    for (int t = 0; t < 8; ++t)
#pragma unroll
        for (int r = 0; r < 8; ++r) acc[t][r] = 0.f;

    for (int s = 0; s < ksteps; ++s) {
        int kq = s * 4;                 // (k0 / 8)
        // A 16x32 f16 layout: halves 0..7 -> K = k0+8*hi+h ; 8..15 -> K = k0+16+8*hi+h
        v8h alo = pA[kq + hi];
        v8h ahi = pA[kq + 2 + hi];
        v16h a = __builtin_shufflevector(alo, ahi,
                 0, 1, 2, 3, 4, 5, 6, 7, 8, 9, 10, 11, 12, 13, 14, 15);
#pragma unroll
        for (int t = 0; t < 8; ++t) {
            const _Float16* Wt = (t < 4) ? WlT : WrT;
            int col = (t & 3) * 16 + l15;
            const v8h* pB = (const v8h*)(Wt + (size_t)col * 64);
            // B 32x16 f16 layout: halves 0..7 -> K = k0+8*hi+h ; 8..15 -> K = k0+16+8*hi+h
            v8h blo = pB[kq + hi];
            v8h bhi = pB[kq + 2 + hi];
            v16h b = __builtin_shufflevector(blo, bhi,
                     0, 1, 2, 3, 4, 5, 6, 7, 8, 9, 10, 11, 12, 13, 14, 15);
            acc[t] = __builtin_amdgcn_wmma_f32_16x16x32_f16(false, a, false, b,
                                                            (short)0, acc[t],
                                                            false, false);
        }
    }

    // C/D layout: VGPR r -> M = r + 8*hi. Padded cols are exact zeros (WhT pad).
    int rbase = wid * 16 + hi * 8;
#pragma unroll
    for (int t = 0; t < 8; ++t) {
        float* Y = (t < 4) ? Yl : Yr;
        int col = (t & 3) * 16 + l15;
#pragma unroll
        for (int r = 0; r < 8; ++r)
            Y[(size_t)(rbase + r) * 64 + col] = acc[t][r];
    }
}

// ---------------- edge kernels (memory/atomic bound) ----------------
__global__ void k_edge_logits(const float* __restrict__ xl, const float* __restrict__ xr,
                              const int* __restrict__ ei, const float* __restrict__ attp,
                              float* __restrict__ aout, unsigned* __restrict__ menc)
{
    int e = blockIdx.x * blockDim.x + threadIdx.x;
    if (e >= ET) return;
    int s, d;
    if (e < EE) { s = ei[e]; d = ei[EE + e]; } else { s = d = e - EE; }
    const float4* pl = (const float4*)(xl + (size_t)s * 64);
    const float4* pr = (const float4*)(xr + (size_t)d * 64);
    const float4* pa = (const float4*)attp;
    float acc = 0.f;
#pragma unroll
    for (int q = 0; q < 16; ++q) {
        float4 l = pl[q], r = pr[q], t = pa[q];
        float v0 = l.x + r.x, v1 = l.y + r.y, v2 = l.z + r.z, v3 = l.w + r.w;
        v0 = v0 > 0.f ? v0 : v0 * SLOPE;
        v1 = v1 > 0.f ? v1 : v1 * SLOPE;
        v2 = v2 > 0.f ? v2 : v2 * SLOPE;
        v3 = v3 > 0.f ? v3 : v3 * SLOPE;
        acc += v0 * t.x + v1 * t.y + v2 * t.z + v3 * t.w;
    }
    aout[e] = acc;
    atomicMax(menc + d, encMax(acc));
}

__global__ void k_edge_weights(float* __restrict__ a, const unsigned* __restrict__ menc,
                               const int* __restrict__ ei, float* __restrict__ denom)
{
    int e = blockIdx.x * blockDim.x + threadIdx.x;
    if (e >= ET) return;
    int d = (e < EE) ? ei[EE + e] : (e - EE);
    float w = __expf(a[e] - decMax(menc[d]));
    a[e] = w;
    atomicAdd(denom + d, w);
}

__global__ void k_edge_aggregate(const float* __restrict__ w, const float* __restrict__ denom,
                                 const int* __restrict__ ei, const float* __restrict__ xl,
                                 float* __restrict__ acc)
{
    int e = blockIdx.x * blockDim.x + threadIdx.x;
    if (e >= ET) return;
    int s, d;
    if (e < EE) { s = ei[e]; d = ei[EE + e]; } else { s = d = e - EE; }
    float alpha = w[e] / denom[d];
    const float4* pl = (const float4*)(xl + (size_t)s * 64);
    float* po = acc + (size_t)d * 64;
#pragma unroll
    for (int q = 0; q < 16; ++q) {
        float4 l = pl[q];
        atomicAdd(po + 4 * q + 0, alpha * l.x);
        atomicAdd(po + 4 * q + 1, alpha * l.y);
        atomicAdd(po + 4 * q + 2, alpha * l.z);
        atomicAdd(po + 4 * q + 3, alpha * l.w);
    }
}

// Adds bias + optional relu in place, and emits the f16 staging copy for the
// next layer's WMMA transform (fused so the f16 conversion costs no extra pass).
__global__ void k_finalize(float* __restrict__ acc, const float* __restrict__ b,
                           _Float16* __restrict__ Xh, int C, int doRelu)
{
    int i = blockIdx.x * blockDim.x + threadIdx.x;
    if (i >= NN * 64) return;
    int c = i & 63;
    float v = acc[i] + ((c < C) ? b[c] : 0.f);
    if (doRelu) v = fmaxf(v, 0.f);
    v = (c < C) ? v : 0.f;      // keep padded columns exactly zero
    acc[i] = v;
    Xh[i]  = (_Float16)v;
}

// ---------------- global mean pool ----------------
__global__ void k_pool_accum(const float* __restrict__ feat, const int* __restrict__ batch,
                             float* __restrict__ pooled, float* __restrict__ counts)
{
    int i = blockIdx.x * blockDim.x + threadIdx.x;
    if (i >= NN * 64) return;
    int n = i >> 6, c = i & 63;
    int g = batch[n];
    if (c == 0) atomicAdd(counts + g, 1.f);
    if (c < OUTC) atomicAdd(pooled + (size_t)g * 64 + c, feat[i]);
}

__global__ void k_pool_final(const float* __restrict__ pooled, const float* __restrict__ counts,
                             float* __restrict__ out)
{
    int i = blockIdx.x * blockDim.x + threadIdx.x;
    if (i >= BB * OUTC) return;
    int g = i / OUTC, c = i % OUTC;
    out[i] = pooled[(size_t)g * 64 + c] / fmaxf(counts[g], 1.f);
}

// ---------------- host launcher ----------------
extern "C" void kernel_launch(void* const* d_in, const int* in_sizes, int n_in,
                              void* d_out, int out_size, void* d_ws, size_t ws_size,
                              hipStream_t stream)
{
    const float* x     = (const float*)d_in[0];
    const int*   ei    = (const int*)d_in[1];
    const int*   batch = (const int*)d_in[2];
    const float* Wl[3]   = {(const float*)d_in[3],  (const float*)d_in[7],  (const float*)d_in[11]};
    const float* Wr[3]   = {(const float*)d_in[4],  (const float*)d_in[8],  (const float*)d_in[12]};
    const float* att[3]  = {(const float*)d_in[5],  (const float*)d_in[9],  (const float*)d_in[13]};
    const float* bias[3] = {(const float*)d_in[6],  (const float*)d_in[10], (const float*)d_in[14]};

    // workspace carve-up (all offsets multiples of 16B)
    char*  base = (char*)d_ws;
    size_t NB   = (size_t)NN * 64 * sizeof(float);           // 25.6 MB
    size_t off  = 0;
    float*    feat  = (float*)(base + off);  off += NB;      // feats / agg accumulator
    float*    xl    = (float*)(base + off);  off += NB;
    float*    xr    = (float*)(base + off);  off += NB;
    float*    ae    = (float*)(base + off);  off += (size_t)ET * 4;
    unsigned* menc  = (unsigned*)(base + off); off += (size_t)NN * 4;
    float*    denom = (float*)(base + off);  off += (size_t)NN * 4;
    float*    attp  = (float*)(base + off);  off += 64 * 4;
    float*    pooled = (float*)(base + off); off += (size_t)BB * 64 * 4;
    float*    counts = (float*)(base + off); off += (size_t)BB * 4;
    _Float16* Xh    = (_Float16*)(base + off); off += (size_t)NN * 64 * 2;  // 12.8 MB
    _Float16* WlT   = (_Float16*)(base + off); off += 64 * 64 * 2;
    _Float16* WrT   = (_Float16*)(base + off); off += 64 * 64 * 2;

    const int rowTiles = NN / 16;                  // 6250 (exact)
    int tBlocks = (rowTiles + 7) / 8;              // 8 waves / 256-thread block
    int eBlocks = (ET + 255) / 256;
    int nBlocks = (NN * 64 + 255) / 256;
    int vBlocks = (NN + 255) / 256;

    struct Layer { int Cin; int Cout; int relu; };
    Layer L[3] = { { INC, HC, 1 }, { HC, HC, 1 }, { HC, OUTC, 0 } };

    // stage layer-0 input once
    k_conv_x0<<<nBlocks, 256, 0, stream>>>(x, Xh);

    for (int l = 0; l < 3; ++l) {
        int ksteps = (L[l].Cin + 31) / 32;         // 1 for layer0, else 2
        k_conv_w<<<(2 * 64 * 64 + 255) / 256, 256, 0, stream>>>(Wl[l], Wr[l], WlT, WrT,
                                                                L[l].Cin, L[l].Cout);
        k_pad_att<<<1, 64, 0, stream>>>(att[l], attp, L[l].Cout);
        gat_transform<<<tBlocks, 256, 0, stream>>>(Xh, WlT, WrT, xl, xr,
                                                   ksteps, rowTiles);
        // raw input dead after transform -> reuse feat as the aggregation accumulator
        k_zero<<<nBlocks, 256, 0, stream>>>(feat, NN * 64);
        k_zero<<<vBlocks, 256, 0, stream>>>((float*)menc, NN);  // 0 < enc(any real)
        k_zero<<<vBlocks, 256, 0, stream>>>(denom, NN);
        k_edge_logits<<<eBlocks, 256, 0, stream>>>(xl, xr, ei, attp, ae, menc);
        k_edge_weights<<<eBlocks, 256, 0, stream>>>(ae, menc, ei, denom);
        k_edge_aggregate<<<eBlocks, 256, 0, stream>>>(ae, denom, ei, xl, feat);
        k_finalize<<<nBlocks, 256, 0, stream>>>(feat, bias[l], Xh, L[l].Cout, L[l].relu);
    }

    k_zero<<<((BB * 64 + BB) + 255) / 256, 256, 0, stream>>>(pooled, BB * 64 + BB);
    k_pool_accum<<<nBlocks, 256, 0, stream>>>(feat, batch, pooled, counts);
    k_pool_final<<<(BB * OUTC + 255) / 256, 256, 0, stream>>>(pooled, counts, (float*)d_out);
}
